// GumbelMultiHeadAttention_60430189855319
// MI455X (gfx1250) — matrix operand order repaired and checked by reading, not compiled, not run on hardware
//
#include <hip/hip_runtime.h>
#include <hip/hip_bf16.h>
#include <stdint.h>

#define BATCH 4
#define SEQ   2048
#define SEQK  2048
#define EMB   512
#define HEADS 8
#define HDIM  64
#define ATT_SCALE 0.125f   // 1/sqrt(64)

typedef __bf16 bf16_t;
typedef __attribute__((ext_vector_type(16))) __bf16 v16bf;
typedef __attribute__((ext_vector_type(8)))  float  v8f;
typedef __attribute__((ext_vector_type(4)))  uint32_t v4u;
typedef __attribute__((ext_vector_type(8)))  int      v8i;
typedef __attribute__((ext_vector_type(4)))  int      v4i;

static __device__ __forceinline__ v8f wmma_bf16(v16bf a, v16bf b, v8f c) {
  // D = A(16x32 bf16) * B(32x16 bf16) + C(16x16 f32)
  return __builtin_amdgcn_wmma_f32_16x16x32_bf16(false, a, false, b, (short)0, c,
                                                 false, false);
}

// Deterministic per-element Gumbel noise (counter-hash stand-in for JAX threefry)
static __device__ __forceinline__ float gumbel_noise(uint32_t idx) {
  uint32_t h = idx * 0x9E3779B9u;
  h ^= h >> 16; h *= 0x85EBCA6Bu;
  h ^= h >> 13; h *= 0xC2B2AE35u;
  h ^= h >> 16;
  float u = ((float)(h >> 8) + 0.5f) * (1.0f / 16777216.0f);  // (0,1)
  return -__logf(-__logf(u));
}

// ---------------------------------------------------------------------------
// TDM: 2D tile DMA  global -> LDS  (D# per CDNA5 ISA §8.3/8.4).
// dims/strides in elements; data_size code 1 = 2-byte elements (bf16).
// LDS byte offset = low 32 bits of the generic pointer (flat-LDS aperture).
// This toolchain exposes the 6-arg builtin:
//   (uint32x4 g0, int32x8 g1, int32x4 g2, int32x4 g3, int32x8, i32 cpol)
// ---------------------------------------------------------------------------
static __device__ __forceinline__ void tdm_load_2d(bf16_t* lds_ptr,
                                                   const bf16_t* gptr,
                                                   uint32_t tensor_d0,
                                                   uint32_t tensor_d1,
                                                   uint32_t tile_d0,
                                                   uint32_t tile_d1,
                                                   uint64_t stride_d0) {
#if __has_builtin(__builtin_amdgcn_tensor_load_to_lds)
  const uint32_t data_size_code = 1u;  // 2-byte elements (bf16)
  const uint64_t ga = (uint64_t)(uintptr_t)gptr;
  v4u g0;
  g0[0] = 1u;                                    // count=1, user descriptor
  g0[1] = (uint32_t)(uintptr_t)lds_ptr;          // lds_addr (bytes)
  g0[2] = (uint32_t)ga;                          // global_addr[31:0]
  g0[3] = (uint32_t)((ga >> 32) & 0x1FFFFFFu) | (2u << 30);  // addr[56:32]|type=2

  // group1 bit layout packed through 64-bit quads
  const uint64_t q0 = ((uint64_t)data_size_code << 16) |      // [17:16] data_size
                      ((uint64_t)tensor_d0 << 48);            // [79:48] lo
  const uint64_t q1 = ((uint64_t)tensor_d0 >> 16) |           // [79:64]
                      ((uint64_t)tensor_d1 << 16) |           // [111:80]
                      ((uint64_t)tile_d0 << 48);              // [127:112]
  const uint64_t q2 = (uint64_t)tile_d1 |                     // [143:128]
                      ((stride_d0 & 0xFFFFFFFFull) << 32);    // [191:160]
  const uint64_t q3 = (stride_d0 >> 32) & 0xFFFFull;          // [207:192]
  v8i g1;
  g1[0] = (int)(uint32_t)q0; g1[1] = (int)(uint32_t)(q0 >> 32);
  g1[2] = (int)(uint32_t)q1; g1[3] = (int)(uint32_t)(q1 >> 32);
  g1[4] = (int)(uint32_t)q2; g1[5] = (int)(uint32_t)(q2 >> 32);
  g1[6] = (int)(uint32_t)q3; g1[7] = (int)(uint32_t)(q3 >> 32);
  v4i gz4 = {0, 0, 0, 0};
  v8i gz8 = {0, 0, 0, 0, 0, 0, 0, 0};
  __builtin_amdgcn_tensor_load_to_lds(g0, g1, gz4, gz4, gz8, 0);
#else
  // Fallback: plain copy by the issuing wave (32 lanes, 8 bf16 per iter/lane)
  const int lane = threadIdx.x & 31;
  const uint32_t total = tile_d0 * tile_d1;
  for (uint32_t e = lane * 8; e < total; e += 32 * 8) {
    const uint32_t r = e / tile_d0, c = e % tile_d0;
    #pragma unroll
    for (int j = 0; j < 8; ++j)
      lds_ptr[e + j] = gptr[(size_t)r * stride_d0 + c + j];
  }
#endif
}

static __device__ __forceinline__ void tdm_wait() {
#if __has_builtin(__builtin_amdgcn_s_wait_tensorcnt)
  __builtin_amdgcn_s_wait_tensorcnt(0);
#endif
}

// ---------------------------------------------------------------------------
// Kernel 1: fused Q/K/V projections.  One wave computes one 16x16 tile of one
// projection.  q,k stored [B,H,L,D] bf16; v stored transposed [B,H,D,Nk] bf16.
// ---------------------------------------------------------------------------
__global__ void qkv_proj_kernel(const float* __restrict__ x,
                                const float* __restrict__ y,
                                const float* __restrict__ z,
                                const float* __restrict__ Wq, const float* __restrict__ bq,
                                const float* __restrict__ Wk, const float* __restrict__ bk,
                                const float* __restrict__ Wv, const float* __restrict__ bv,
                                bf16_t* __restrict__ qbuf,
                                bf16_t* __restrict__ kbuf,
                                bf16_t* __restrict__ vtbuf) {
  const int lane = threadIdx.x & 31;
  const int wv   = threadIdx.x >> 5;
  const int W    = blockIdx.x * 8 + wv;      // 0 .. 49151
  const int p    = W / 16384;                // 0=q, 1=k, 2=v
  const int t    = W % 16384;
  const int row0 = (t >> 5) << 4;            // 512 row tiles (M = 8192)
  const int col0 = (t & 31) << 4;            // 32 col tiles  (N = 512)
  const int ha   = lane >> 4;
  const int ln   = lane & 15;

  const float* A    = (p == 0) ? x  : (p == 1) ? y  : z;
  const float* Wm   = (p == 0) ? Wq : (p == 1) ? Wk : Wv;
  const float* bias = (p == 0) ? bq : (p == 1) ? bk : bv;

  const float* ap_base = A  + (size_t)(row0 + ln) * EMB;  // A row for this lane
  const float* bp_base = Wm + (size_t)(col0 + ln) * EMB;  // W row = output col

  v8f acc = {};
  #pragma unroll 4
  for (int k0 = 0; k0 < EMB; k0 += 32) {
    __builtin_prefetch(bp_base + k0 + 256, 0, 3);
    const float* ap = ap_base + k0 + ha * 16;  // A layout: K = ha*16 + 0..15
    const float* bp = bp_base + k0 + ha * 16;  // B layout: K = ha*16 + 0..15
    v16bf a, b;
    #pragma unroll
    for (int j = 0; j < 8; ++j) { a[j] = (bf16_t)ap[j]; a[j + 8] = (bf16_t)ap[8 + j]; }
    #pragma unroll
    for (int j = 0; j < 16; ++j) b[j] = (bf16_t)bp[j];
    acc = wmma_bf16(a, b, acc);
  }

  const int   ncol = col0 + ln;
  const int   h    = ncol >> 6, d = ncol & 63;
  const float bb   = bias[ncol];
  #pragma unroll
  for (int v = 0; v < 8; ++v) {
    const int   R  = row0 + ha * 8 + v;        // global row in [0, B*SEQ)
    const int   b_ = R >> 11, nn = R & (SEQ - 1);
    const float val = acc[v] + bb;
    if (p == 0)
      qbuf[(((size_t)b_ * HEADS + h) * SEQ + nn) * HDIM + d] = (bf16_t)val;
    else if (p == 1)
      kbuf[(((size_t)b_ * HEADS + h) * SEQK + nn) * HDIM + d] = (bf16_t)val;
    else
      vtbuf[(((size_t)b_ * HEADS + h) * HDIM + d) * SEQK + nn] = (bf16_t)val;
  }
}

// ---------------------------------------------------------------------------
// Kernel 2: flash attention with Gumbel-softmax.
// One block = one (b,h) x 128 q rows (8 waves x 16 rows).  K/V chunks (32 keys)
// are staged into LDS by the Tensor Data Mover, double-buffered, and shared by
// all 8 waves.  Output [B,N,C] bf16.
// ---------------------------------------------------------------------------
__global__ void attn_kernel(const bf16_t* __restrict__ qbuf,
                            const bf16_t* __restrict__ kbuf,
                            const bf16_t* __restrict__ vtbuf,
                            bf16_t* __restrict__ obuf) {
  __shared__ bf16_t kL[2][32 * HDIM];   // [buf][key][d]   4 KB each
  __shared__ bf16_t vL[2][HDIM * 32];   // [buf][d][key]   4 KB each
  __shared__ bf16_t psm[8][16 * 32];    // per-wave P staging, 8 KB

  const int lane = threadIdx.x & 31;
  const int wv   = threadIdx.x >> 5;
  const int bh   = blockIdx.x >> 4;              // (b*H + h)
  const int row0 = ((blockIdx.x & 15) << 7) + (wv << 4);
  const int b_   = bh / HEADS, h = bh % HEADS;
  const int ha   = lane >> 4, ln = lane & 15;

  const bf16_t* kG = kbuf  + (size_t)bh * SEQK * HDIM;   // [Nk][64]
  const bf16_t* vG = vtbuf + (size_t)bh * HDIM * SEQK;   // [64][Nk]

  // Q tile as two A fragments (K = D = 64 -> 2 x 32)
  v16bf aq[2];
  {
    const bf16_t* qrow = qbuf + ((size_t)bh * SEQ + row0 + ln) * HDIM;
    #pragma unroll
    for (int w = 0; w < 2; ++w) {
      const bf16_t* qp = qrow + w * 32 + ha * 16;
      #pragma unroll
      for (int j = 0; j < 8; ++j) { aq[w][j] = qp[j]; aq[w][j + 8] = qp[8 + j]; }
    }
  }

  float m_run[8], l_run[8];
  v8f   acc_o[4] = {{}, {}, {}, {}};             // 16 x 64 f32 accumulator
  #pragma unroll
  for (int v = 0; v < 8; ++v) { m_run[v] = -3.0e38f; l_run[v] = 0.0f; }

  const uint32_t idx_base = ((uint32_t)bh * SEQ + (uint32_t)row0) * (uint32_t)SEQK;
  bf16_t* pw = psm[wv];
  const int NCHUNK = SEQK / 32;                  // 64

  // prologue: DMA chunk 0 into buffer 0
  if (wv == 0) {
    tdm_load_2d(kL[0], kG, HDIM, 32, HDIM, 32, HDIM);
    tdm_load_2d(vL[0], vG, SEQK, HDIM, 32, HDIM, SEQK);
    tdm_wait();
  }
  __syncthreads();

  for (int i = 0; i < NCHUNK; ++i) {
    const int kc  = i * 32;
    const int buf = i & 1;

    // async prefetch of next chunk into the other buffer
    if (wv == 0 && i + 1 < NCHUNK) {
      const int kn = kc + 32;
      tdm_load_2d(kL[buf ^ 1], kG + (size_t)kn * HDIM, HDIM, 32, HDIM, 32, HDIM);
      tdm_load_2d(vL[buf ^ 1], vG + kn, SEQK, HDIM, 32, HDIM, SEQK);
    }

    // --- logits: two 16x16 C tiles covering keys kc..kc+31 (from LDS) ---
    v8f s0 = {}, s1 = {};
    #pragma unroll
    for (int w = 0; w < 2; ++w) {
      const bf16_t* kb = kL[buf] + ln * HDIM + w * 32 + ha * 16;
      v16bf bk0 = *(const v16bf*)kb;                    // keys kc+ln
      v16bf bk1 = *(const v16bf*)(kb + 16 * HDIM);      // keys kc+16+ln
      s0 = wmma_bf16(aq[w], bk0, s0);
      s1 = wmma_bf16(aq[w], bk1, s1);
    }

    // --- scale + gumbel + online softmax (row = ha*8+v, key axis across lanes)
    float p0[8], p1[8];
    #pragma unroll
    for (int v = 0; v < 8; ++v) {
      const uint32_t ir = idx_base + (uint32_t)(ha * 8 + v) * SEQK +
                          (uint32_t)kc + (uint32_t)ln;
      s0[v] = s0[v] * ATT_SCALE + gumbel_noise(ir);
      s1[v] = s1[v] * ATT_SCALE + gumbel_noise(ir + 16u);
      float mx = fmaxf(s0[v], s1[v]);
      #pragma unroll
      for (int off = 8; off >= 1; off >>= 1) mx = fmaxf(mx, __shfl_xor(mx, off, 32));
      const float mnew  = fmaxf(m_run[v], mx);
      const float alpha = __expf(m_run[v] - mnew);
      m_run[v] = mnew;
      p0[v] = __expf(s0[v] - mnew);
      p1[v] = __expf(s1[v] - mnew);
      float rs = p0[v] + p1[v];
      #pragma unroll
      for (int off = 8; off >= 1; off >>= 1) rs += __shfl_xor(rs, off, 32);
      l_run[v] = l_run[v] * alpha + rs;
      #pragma unroll
      for (int td = 0; td < 4; ++td) acc_o[td][v] *= alpha;
    }

    // --- C-layout P -> LDS -> A-layout bf16 fragment (16 rows x 32 keys) ---
    #pragma unroll
    for (int v = 0; v < 8; ++v) {
      pw[(ha * 8 + v) * 32 + ln]      = (bf16_t)p0[v];
      pw[(ha * 8 + v) * 32 + 16 + ln] = (bf16_t)p1[v];
    }
    __asm__ volatile("s_wait_dscnt 0" ::: "memory");
    v16bf a_p;
    #pragma unroll
    for (int j = 0; j < 8; ++j) {
      a_p[j]     = pw[ln * 32 + ha * 16 + j];
      a_p[j + 8] = pw[ln * 32 + ha * 16 + 8 + j];
    }

    // --- P (16x32) @ V chunk (32 x 64) from LDS (V is d-major) ---
    #pragma unroll
    for (int td = 0; td < 4; ++td) {
      const bf16_t* vp = vL[buf] + (td * 16 + ln) * 32 + ha * 16;
      v16bf bvv = *(const v16bf*)vp;
      acc_o[td] = wmma_bf16(a_p, bvv, acc_o[td]);
    }

    // make next chunk's LDS data visible to everyone before buffer swap
    if (wv == 0 && i + 1 < NCHUNK) tdm_wait();
    __syncthreads();
  }

  // normalize and store to [B,N,C] bf16
  #pragma unroll
  for (int td = 0; td < 4; ++td) {
    const int d = td * 16 + ln;
    #pragma unroll
    for (int v = 0; v < 8; ++v) {
      const int   row = row0 + ha * 8 + v;
      const float o   = acc_o[td][v] / l_run[v];
      obuf[((size_t)b_ * SEQ + row) * EMB + h * HDIM + d] = (bf16_t)o;
    }
  }
}

// ---------------------------------------------------------------------------
// Kernel 3: output projection  out = O @ Wo^T + bo  (f32 output)
// ---------------------------------------------------------------------------
__global__ void out_proj_kernel(const bf16_t* __restrict__ obuf,
                                const float* __restrict__ Wo,
                                const float* __restrict__ bo,
                                float* __restrict__ out) {
  const int lane = threadIdx.x & 31;
  const int wv   = threadIdx.x >> 5;
  const int W    = blockIdx.x * 8 + wv;      // 0 .. 16383
  const int row0 = (W >> 5) << 4;
  const int col0 = (W & 31) << 4;
  const int ha   = lane >> 4, ln = lane & 15;

  const bf16_t* ap_base = obuf + (size_t)(row0 + ln) * EMB;
  const float*  bp_base = Wo   + (size_t)(col0 + ln) * EMB;

  v8f acc = {};
  #pragma unroll 4
  for (int k0 = 0; k0 < EMB; k0 += 32) {
    __builtin_prefetch(bp_base + k0 + 256, 0, 3);
    const bf16_t* ap = ap_base + k0 + ha * 16;
    const float*  bp = bp_base + k0 + ha * 16;
    v16bf a, b;
    #pragma unroll
    for (int j = 0; j < 8; ++j) { a[j] = ap[j]; a[j + 8] = ap[8 + j]; }
    #pragma unroll
    for (int j = 0; j < 16; ++j) b[j] = (bf16_t)bp[j];
    acc = wmma_bf16(a, b, acc);
  }

  const int   ncol = col0 + ln;
  const float bb   = bo[ncol];
  #pragma unroll
  for (int v = 0; v < 8; ++v)
    out[(size_t)(row0 + ha * 8 + v) * EMB + ncol] = acc[v] + bb;
}

// ---------------------------------------------------------------------------
extern "C" void kernel_launch(void* const* d_in, const int* in_sizes, int n_in,
                              void* d_out, int out_size, void* d_ws, size_t ws_size,
                              hipStream_t stream) {
  (void)in_sizes; (void)n_in; (void)out_size; (void)ws_size;
  const float* x  = (const float*)d_in[0];
  const float* y  = (const float*)d_in[1];
  const float* z  = (const float*)d_in[2];
  const float* Wq = (const float*)d_in[3];
  const float* bq = (const float*)d_in[4];
  const float* Wk = (const float*)d_in[5];
  const float* bk = (const float*)d_in[6];
  const float* Wv = (const float*)d_in[7];
  const float* bv = (const float*)d_in[8];
  const float* Wo = (const float*)d_in[9];
  const float* bo = (const float*)d_in[10];

  const size_t qkvElems = (size_t)BATCH * HEADS * SEQ * HDIM;  // 4M bf16 each
  bf16_t* qbuf  = (bf16_t*)d_ws;
  bf16_t* kbuf  = qbuf  + qkvElems;
  bf16_t* vtbuf = kbuf  + qkvElems;
  bf16_t* obuf  = vtbuf + qkvElems;           // [B,N,C] bf16 (4M elems)

  // 3 * 16384 wave-tiles / 8 waves per block
  qkv_proj_kernel<<<6144, 256, 0, stream>>>(x, y, z, Wq, bq, Wk, bk, Wv, bv,
                                            qbuf, kbuf, vtbuf);
  // one block per (b,h) x 128 q rows: 32 * 16 = 512 blocks
  attn_kernel<<<512, 256, 0, stream>>>(qbuf, kbuf, vtbuf, obuf);
  // 16384 wave-tiles / 8
  out_proj_kernel<<<2048, 256, 0, stream>>>(obuf, Wo, bo, (float*)d_out);
}
